// MSDeformAttnTransformerEncoder_63668595196599
// MI455X (gfx1250) — matrix operand order repaired and checked by reading, not compiled, not run on hardware
//
#include <hip/hip_runtime.h>
#include <hip/hip_bf16.h>
#include <cstdint>
#include <cstddef>

// ---------------------------------------------------------------------------
// MS-Deformable-Attention Transformer Encoder for MI455X (gfx1250, wave32)
// GEMMs: v_wmma_f32_16x16x32_bf16, double-buffered LDS, pre-transposed bf16 W.
// Staging uses GLOBAL_LOAD_ASYNC_TO_LDS_B128 (ASYNCcnt) when the toolchain
// exposes the builtin; otherwise falls back to reg-staged b128 copies.
// ---------------------------------------------------------------------------

typedef __bf16 bf16;
typedef __attribute__((ext_vector_type(16))) __bf16 v16bf;
typedef __attribute__((ext_vector_type(8)))  float  v8f;
typedef __attribute__((ext_vector_type(4)))  unsigned int u32x4;
typedef __attribute__((ext_vector_type(4)))  int i32x4;

#define B_   2
#define C_   256
#define NH_  8
#define DH_  32
#define LV_  4
#define P_   4
#define F_   1024
#define L_   6
#define NT_  21760              // 128*128 + 64*64 + 32*32 + 16*16
#define MN_  (B_ * NT_)         // 43520 rows (multiple of 128)

#if defined(__has_builtin)
#  if __has_builtin(__builtin_amdgcn_global_load_async_to_lds_b128)
#    define ASYNC_LDS 1
#  endif
#endif
#ifndef ASYNC_LDS
#  define ASYNC_LDS 0
#endif

#if ASYNC_LDS
#  if __has_builtin(__builtin_amdgcn_s_wait_asynccnt)
#    define WAIT_ASYNC() __builtin_amdgcn_s_wait_asynccnt(0)
#  else
#    define WAIT_ASYNC() asm volatile("s_wait_asynccnt 0" ::: "memory")
#  endif
// builtin signature (from probe): (int4 AS1*, int4 AS3*, imm int, imm int)
typedef __attribute__((address_space(1))) i32x4* gp128;
typedef __attribute__((address_space(3))) i32x4* lp128;
#endif

static __device__ __forceinline__ v8f v8f_zero() {
  v8f z;
#pragma unroll
  for (int i = 0; i < 8; ++i) z[i] = 0.0f;
  return z;
}

// ---------------------------------------------------------------------------
// Pack one pyramid level: (B,C,H,W) -> x[b, start+hw, c]; pos += level_embed
// ---------------------------------------------------------------------------
__global__ void pack_level_kernel(const float* __restrict__ src,
                                  const float* __restrict__ pos_in,
                                  const float* __restrict__ lvl_emb, // + li*C
                                  float* __restrict__ x,
                                  float* __restrict__ pos,
                                  int start, int HW) {
  size_t i = (size_t)blockIdx.x * blockDim.x + threadIdx.x;
  size_t total = (size_t)B_ * C_ * HW;
  if (i >= total) return;
  int hw = (int)(i % HW);
  int c  = (int)((i / HW) % C_);
  int b  = (int)(i / ((size_t)HW * C_));
  size_t o = ((size_t)b * NT_ + start + hw) * C_ + c;
  x[o]   = src[i];
  pos[o] = pos_in[i] + lvl_emb[c];
}

// ---------------------------------------------------------------------------
// Weight prep: fp32 W[K][Nc] -> bf16 WT[Nc][K], LDS-tiled 32x32 transpose.
// ---------------------------------------------------------------------------
__global__ __launch_bounds__(256) void wprep_kernel(
    const float* __restrict__ W, bf16* __restrict__ WT, int K, int Nc) {
  __shared__ bf16 tile[32][34];
  const size_t lo = (size_t)blockIdx.z * K * Nc;
  const int nb = blockIdx.x * 32, kb = blockIdx.y * 32;
  const int tx = threadIdx.x & 31, ty = threadIdx.x >> 5;   // ty: 0..7
#pragma unroll
  for (int j = 0; j < 4; ++j) {
    int kl = ty + 8 * j;
    tile[kl][tx] = (bf16)W[lo + (size_t)(kb + kl) * Nc + nb + tx];
  }
  __syncthreads();
#pragma unroll
  for (int j = 0; j < 4; ++j) {
    int nl = ty + 8 * j;
    WT[lo + (size_t)(nb + nl) * K + kb + tx] = tile[tx][nl];
  }
}

// ---------------------------------------------------------------------------
// q = bf16(x + pos), x16 = bf16(x)
// ---------------------------------------------------------------------------
__global__ void qmake_kernel(const float* __restrict__ x,
                             const float* __restrict__ pos,
                             bf16* __restrict__ q16,
                             bf16* __restrict__ x16,
                             size_t total) {
  size_t i = (size_t)blockIdx.x * blockDim.x + threadIdx.x;
  if (i >= total) return;
  float xv = x[i];
  q16[i] = (bf16)(xv + pos[i]);
  x16[i] = (bf16)xv;
}

// ---------------------------------------------------------------------------
// Tiled WMMA GEMM: D[M,Nc] = A[M,K](bf16) @ WT[Nc,K](bf16)^T + bias (+Res)(ReLU)
// Block 128x64, 8 waves, wave tile 32x32 (2x2 WMMA). KT=64, double-buffered
// LDS, one barrier per K-tile. M%128==0, K%64==0, Nc%64==0 for all uses.
// ---------------------------------------------------------------------------
template <int RELU, int RES, int OUTB>
__global__ __launch_bounds__(256) void gemm_wmma_kernel(
    const bf16*  __restrict__ A,
    const bf16*  __restrict__ WT,
    const float* __restrict__ bias,
    const float* __restrict__ Res,
    float*       __restrict__ outF,
    bf16*        __restrict__ outB,
    int M, int K, int Nc) {
  constexpr int KT  = 64;
  constexpr int LDA = 72;   // 64 + 8 pad (bf16) -> 144B row stride, 16B aligned
  constexpr int LDB = 72;
  __shared__ __align__(16) bf16 sA[2][128 * LDA];
  __shared__ __align__(16) bf16 sB[2][64 * LDB];

  const int tid  = threadIdx.x;
  const int lane = tid & 31;
  const int wave = tid >> 5;
  const int wm = (wave & 3) * 32, wn = (wave >> 2) * 32;
  const int m0 = blockIdx.y * 128, n0 = blockIdx.x * 64;

  // staging geometry: A tile 128x64 = 1024 x 16B segs (4/thread, one row half),
  //                   B tile  64x64 =  512 x 16B segs (2/thread)
  const int arow = tid >> 1, aseg = (tid & 1) * 4;
  const int brow = tid >> 2, bseg = (tid & 3) * 2;
  const bf16* aptr = A  + (size_t)(m0 + arow) * K + aseg * 8;
  const bf16* bptr = WT + (size_t)(n0 + brow) * K + bseg * 8;

  v8f acc[2][2];
#pragma unroll
  for (int i = 0; i < 2; ++i)
#pragma unroll
    for (int j = 0; j < 2; ++j) acc[i][j] = v8f_zero();

  const int lcol  = lane & 15;
  const int ahalf = (lane < 16) ? 0 : 8;    // A frag: lanes 0-15 K {0..7,16..23}
  const int bhalf = (lane < 16) ? 0 : 8;    // B frag: lanes 0-15 K {0..7,16..23}

  auto compute = [&](int buf) {
#pragma unroll
    for (int ks = 0; ks < 2; ++ks) {          // two 32-deep WMMA steps per tile
      v16bf afrag[2], bfrag[2];
#pragma unroll
      for (int i = 0; i < 2; ++i) {
        int r = wm + i * 16 + lcol;
        u32x4* fp = reinterpret_cast<u32x4*>(&afrag[i]);
        fp[0] = *reinterpret_cast<const u32x4*>(&sA[buf][r * LDA + ks * 32 + ahalf]);
        fp[1] = *reinterpret_cast<const u32x4*>(&sA[buf][r * LDA + ks * 32 + 16 + ahalf]);
      }
#pragma unroll
      for (int j = 0; j < 2; ++j) {
        int c = wn + j * 16 + lcol;
        u32x4* fp = reinterpret_cast<u32x4*>(&bfrag[j]);
        fp[0] = *reinterpret_cast<const u32x4*>(&sB[buf][c * LDB + ks * 32 + bhalf]);
        fp[1] = *reinterpret_cast<const u32x4*>(&sB[buf][c * LDB + ks * 32 + 16 + bhalf]);
      }
#pragma unroll
      for (int i = 0; i < 2; ++i)
#pragma unroll
        for (int j = 0; j < 2; ++j)
          acc[i][j] = __builtin_amdgcn_wmma_f32_16x16x32_bf16(
              false, afrag[i], false, bfrag[j], (short)0, acc[i][j], false, false);
    }
  };

  const int T = K / KT;
  int cur = 0;

#if ASYNC_LDS
  // --- async DMA staging: memory -> LDS without VGPR round-trip (ASYNCcnt) ---
  auto stage_async = [&](int k0, int buf) {
#pragma unroll
    for (int s = 0; s < 4; ++s)
      __builtin_amdgcn_global_load_async_to_lds_b128(
          (gp128)(void*)const_cast<bf16*>(aptr + k0 + s * 8),
          (lp128)(void*)&sA[buf][arow * LDA + (aseg + s) * 8], 0, 0);
#pragma unroll
    for (int s = 0; s < 2; ++s)
      __builtin_amdgcn_global_load_async_to_lds_b128(
          (gp128)(void*)const_cast<bf16*>(bptr + k0 + s * 8),
          (lp128)(void*)&sB[buf][brow * LDB + (bseg + s) * 8], 0, 0);
  };
  stage_async(0, 0);
  for (int t = 0; t < T; ++t) {
    WAIT_ASYNC();                              // wave's copies into lds[cur] done
    __syncthreads();                           // all waves' copies done
    if (t + 1 < T) stage_async((t + 1) * KT, cur ^ 1);  // overlap with compute
    compute(cur);
    cur ^= 1;
  }
#else
  // --- fallback: register-staged b128 copies ---
  u32x4 areg[4], breg[2];
  auto load_tile = [&](int k0) {
#pragma unroll
    for (int s = 0; s < 4; ++s)
      areg[s] = *reinterpret_cast<const u32x4*>(aptr + k0 + s * 8);
#pragma unroll
    for (int s = 0; s < 2; ++s)
      breg[s] = *reinterpret_cast<const u32x4*>(bptr + k0 + s * 8);
  };
  auto store_tile = [&](int buf) {
#pragma unroll
    for (int s = 0; s < 4; ++s)
      *reinterpret_cast<u32x4*>(&sA[buf][arow * LDA + (aseg + s) * 8]) = areg[s];
#pragma unroll
    for (int s = 0; s < 2; ++s)
      *reinterpret_cast<u32x4*>(&sB[buf][brow * LDB + (bseg + s) * 8]) = breg[s];
  };
  load_tile(0);
  for (int t = 0; t < T; ++t) {
    store_tile(cur);
    if (t + 1 < T) load_tile((t + 1) * KT);
    __syncthreads();
    compute(cur);
    cur ^= 1;
  }
#endif

  // epilogue: C/D layout — col = lane%16, row = v + 8*(lane>=16)
#pragma unroll
  for (int i = 0; i < 2; ++i) {
#pragma unroll
    for (int j = 0; j < 2; ++j) {
      int col  = n0 + wn + j * 16 + lcol;
      float bv = bias[col];
#pragma unroll
      for (int v = 0; v < 8; ++v) {
        int row = m0 + wm + i * 16 + v + ahalf;
        size_t idx = (size_t)row * Nc + col;
        float val = acc[i][j][v] + bv;
        if (RES)  val += Res[idx];
        if (RELU) val = fmaxf(val, 0.0f);
        if (OUTB) outB[idx] = (bf16)val;
        else      outF[idx] = val;
      }
    }
  }
}

// ---------------------------------------------------------------------------
// Softmax over groups of 16 (per b,n,head). attn layout (B*N, NH*16) -> t*16.
// ---------------------------------------------------------------------------
__global__ void softmax16_kernel(float* __restrict__ a, int total) {
  int t = blockIdx.x * blockDim.x + threadIdx.x;
  if (t >= total) return;
  float* p = a + (size_t)t * 16;
  float mx = -1e30f;
#pragma unroll
  for (int i = 0; i < 16; ++i) mx = fmaxf(mx, p[i]);
  float e[16], s = 0.0f;
#pragma unroll
  for (int i = 0; i < 16; ++i) { e[i] = __expf(p[i] - mx); s += e[i]; }
  float inv = 1.0f / s;
#pragma unroll
  for (int i = 0; i < 16; ++i) p[i] = e[i] * inv;
}

// ---------------------------------------------------------------------------
// Deformable bilinear sampler: one wave32 per (b,n,head); lane = channel d.
// ---------------------------------------------------------------------------
__global__ __launch_bounds__(256) void msdeform_sample_kernel(
    const float* __restrict__ value,
    const float* __restrict__ off,
    const float* __restrict__ attn,
    bf16* __restrict__ outB) {
  int gw   = blockIdx.x * 8 + (threadIdx.x >> 5);
  int lane = threadIdx.x & 31;
  int hh    = gw & (NH_ - 1);
  int token = gw >> 3;
  int n = token % NT_;
  int b = token / NT_;

  int loc0, hq, wq;
  if      (n < 16384) { loc0 = n;         hq = 128; wq = 128; }
  else if (n < 20480) { loc0 = n - 16384; hq = 64;  wq = 64;  }
  else if (n < 21504) { loc0 = n - 20480; hq = 32;  wq = 32;  }
  else                { loc0 = n - 21504; hq = 16;  wq = 16;  }
  int rr = loc0 / wq, cc = loc0 % wq;
  float refx = (cc + 0.5f) / (float)wq;
  float refy = (rr + 0.5f) / (float)hq;

  const size_t bn      = (size_t)b * NT_ + n;
  const size_t offbase = (bn * NH_ + hh) * (LV_ * P_ * 2);
  const size_t awbase  = (bn * NH_ + hh) * (LV_ * P_);

  const int starts[4] = {0, 16384, 20480, 21504};
  const int dims[4]   = {128, 64, 32, 16};

  float accv = 0.0f;
#pragma unroll
  for (int lv = 0; lv < LV_; ++lv) {
    const int   di = dims[lv];
    const float dl = (float)di;
    const float* vb = value + ((size_t)b * NT_ + starts[lv]) * C_ + hh * DH_ + lane;
#pragma unroll
    for (int p = 0; p < P_; ++p) {
      float ox = off[offbase + (lv * P_ + p) * 2 + 0];
      float oy = off[offbase + (lv * P_ + p) * 2 + 1];
      float aw = attn[awbase + lv * P_ + p];
      float sx = (refx + ox / dl) * dl - 0.5f;
      float sy = (refy + oy / dl) * dl - 0.5f;
      float fx = floorf(sx), fy = floorf(sy);
      float tx = sx - fx, ty = sy - fy;
      int x0 = (int)fx, y0 = (int)fy;

      auto fetch = [&](int xi, int yi) -> float {
        if (xi < 0 || xi >= di || yi < 0 || yi >= di) return 0.0f;
        return vb[(size_t)(yi * di + xi) * C_];
      };
      float s = fetch(x0,     y0    ) * (1.0f - tx) * (1.0f - ty)
              + fetch(x0 + 1, y0    ) * tx          * (1.0f - ty)
              + fetch(x0,     y0 + 1) * (1.0f - tx) * ty
              + fetch(x0 + 1, y0 + 1) * tx          * ty;
      accv += aw * s;
    }
  }
  outB[bn * C_ + hh * DH_ + lane] = (bf16)accv;
}

// ---------------------------------------------------------------------------
// LayerNorm over C=256: one wave32 per row, 8 cols per lane, shfl reduction.
// ---------------------------------------------------------------------------
__global__ __launch_bounds__(256) void layernorm_kernel(
    const float* __restrict__ in,
    const float* __restrict__ g,
    const float* __restrict__ bta,
    float* __restrict__ outF,
    bf16*  __restrict__ outB,
    int rows) {
  int r    = blockIdx.x * 8 + (threadIdx.x >> 5);
  int lane = threadIdx.x & 31;
  if (r >= rows) return;
  const float* row = in + (size_t)r * C_;
  float v[8], s = 0.0f, s2 = 0.0f;
#pragma unroll
  for (int j = 0; j < 8; ++j) {
    v[j] = row[lane + j * 32];
    s  += v[j];
    s2 += v[j] * v[j];
  }
#pragma unroll
  for (int m = 16; m >= 1; m >>= 1) {
    s  += __shfl_xor(s,  m, 32);
    s2 += __shfl_xor(s2, m, 32);
  }
  float mu  = s  * (1.0f / C_);
  float var = s2 * (1.0f / C_) - mu * mu;
  float rs  = rsqrtf(var + 1e-5f);
#pragma unroll
  for (int j = 0; j < 8; ++j) {
    int c = lane + j * 32;
    float y = (v[j] - mu) * rs * g[c] + bta[c];
    outF[(size_t)r * C_ + c] = y;
    if (outB) outB[(size_t)r * C_ + c] = (bf16)y;
  }
}

// ---------------------------------------------------------------------------
// Host launcher
// ---------------------------------------------------------------------------
extern "C" void kernel_launch(void* const* d_in, const int* in_sizes, int n_in,
                              void* d_out, int out_size, void* d_ws, size_t ws_size,
                              hipStream_t stream) {
  (void)in_sizes; (void)n_in; (void)out_size; (void)ws_size;
  const float* srcp[4] = {(const float*)d_in[0], (const float*)d_in[2],
                          (const float*)d_in[4], (const float*)d_in[6]};
  const float* posp[4] = {(const float*)d_in[1], (const float*)d_in[3],
                          (const float*)d_in[5], (const float*)d_in[7]};
  const float* lvl_emb = (const float*)d_in[8];
  const float* W_off   = (const float*)d_in[9];
  const float* b_off   = (const float*)d_in[10];
  const float* W_attn  = (const float*)d_in[11];
  const float* b_attn  = (const float*)d_in[12];
  const float* W_val   = (const float*)d_in[13];
  const float* b_val   = (const float*)d_in[14];
  const float* W_out   = (const float*)d_in[15];
  const float* b_out   = (const float*)d_in[16];
  const float* ln1_g   = (const float*)d_in[17];
  const float* ln1_b   = (const float*)d_in[18];
  const float* W_ff1   = (const float*)d_in[19];
  const float* b_ff1   = (const float*)d_in[20];
  const float* W_ff2   = (const float*)d_in[21];
  const float* b_ff2   = (const float*)d_in[22];
  const float* ln2_g   = (const float*)d_in[23];
  const float* ln2_b   = (const float*)d_in[24];

  const size_t MNC = (size_t)MN_ * C_;   // 11,141,120

  char* p = (char*)d_ws;
  auto carve = [&](size_t bytes) -> char* {
    char* q = p;
    p += (bytes + 255) & ~(size_t)255;
    return q;
  };
  float* x     = (float*)carve(MNC * 4);
  float* pos   = (float*)carve(MNC * 4);
  float* value = (float*)carve(MNC * 4);
  float* tmp   = (float*)carve(MNC * 4);
  float* attn  = (float*)carve((size_t)MN_ * 128 * 4);
  bf16*  x16   = (bf16*)carve(MNC * 2);
  bf16*  q16   = (bf16*)carve(MNC * 2);
  bf16*  a16   = (bf16*)carve(MNC * 2);
  bf16*  h16   = (bf16*)carve((size_t)MN_ * F_ * 2);
  bf16* wt_val  = (bf16*)carve((size_t)L_ * C_ * C_ * 2);
  bf16* wt_off  = (bf16*)carve((size_t)L_ * C_ * 256 * 2);
  bf16* wt_attn = (bf16*)carve((size_t)L_ * C_ * 128 * 2);
  bf16* wt_out  = (bf16*)carve((size_t)L_ * C_ * C_ * 2);
  bf16* wt_ff1  = (bf16*)carve((size_t)L_ * C_ * F_ * 2);
  bf16* wt_ff2  = (bf16*)carve((size_t)L_ * F_ * C_ * 2);

  const dim3 blk(256);

  // ---- weight prep: fp32 [K][Nc] -> bf16 [Nc][K], all layers at once ----
  wprep_kernel<<<dim3(C_ / 32,  C_ / 32, L_), blk, 0, stream>>>(W_val,  wt_val,  C_, C_);
  wprep_kernel<<<dim3(256 / 32, C_ / 32, L_), blk, 0, stream>>>(W_off,  wt_off,  C_, 256);
  wprep_kernel<<<dim3(128 / 32, C_ / 32, L_), blk, 0, stream>>>(W_attn, wt_attn, C_, 128);
  wprep_kernel<<<dim3(C_ / 32,  C_ / 32, L_), blk, 0, stream>>>(W_out,  wt_out,  C_, C_);
  wprep_kernel<<<dim3(F_ / 32,  C_ / 32, L_), blk, 0, stream>>>(W_ff1,  wt_ff1,  C_, F_);
  wprep_kernel<<<dim3(C_ / 32,  F_ / 32, L_), blk, 0, stream>>>(W_ff2,  wt_ff2,  F_, C_);

  // ---- pack pyramid levels into (B,N,C) token layout ----
  {
    const int starts[4] = {0, 16384, 20480, 21504};
    const int hw[4]     = {16384, 4096, 1024, 256};
    for (int li = 0; li < 4; ++li) {
      size_t total = (size_t)B_ * C_ * hw[li];
      pack_level_kernel<<<dim3((total + 255) / 256), blk, 0, stream>>>(
          srcp[li], posp[li], lvl_emb + li * C_, x, pos, starts[li], hw[li]);
    }
  }

  for (int l = 0; l < L_; ++l) {
    const bf16* Wv = wt_val  + (size_t)l * C_ * C_;
    const bf16* Wo = wt_off  + (size_t)l * C_ * 256;
    const bf16* Wa = wt_attn + (size_t)l * C_ * 128;
    const bf16* Wu = wt_out  + (size_t)l * C_ * C_;
    const bf16* W1 = wt_ff1  + (size_t)l * C_ * F_;
    const bf16* W2 = wt_ff2  + (size_t)l * F_ * C_;

    qmake_kernel<<<dim3((MNC + 255) / 256), blk, 0, stream>>>(x, pos, q16, x16, MNC);

    // value = x16 @ W_val + b_val
    gemm_wmma_kernel<0, 0, 0><<<dim3(C_ / 64, MN_ / 128), blk, 0, stream>>>(
        x16, Wv, b_val + l * C_, nullptr, value, nullptr, MN_, C_, C_);
    // off = q16 @ W_off + b_off
    gemm_wmma_kernel<0, 0, 0><<<dim3(256 / 64, MN_ / 128), blk, 0, stream>>>(
        q16, Wo, b_off + l * 256, nullptr, tmp, nullptr, MN_, C_, 256);
    // attn = q16 @ W_attn + b_attn
    gemm_wmma_kernel<0, 0, 0><<<dim3(128 / 64, MN_ / 128), blk, 0, stream>>>(
        q16, Wa, b_attn + l * 128, nullptr, attn, nullptr, MN_, C_, 128);

    softmax16_kernel<<<dim3((MN_ * NH_ + 255) / 256), blk, 0, stream>>>(attn, MN_ * NH_);

    msdeform_sample_kernel<<<dim3(MN_ * NH_ / 8), blk, 0, stream>>>(value, tmp, attn, a16);

    // out-proj + residual: tmp = a16 @ W_out + b_out + x
    gemm_wmma_kernel<0, 1, 0><<<dim3(C_ / 64, MN_ / 128), blk, 0, stream>>>(
        a16, Wu, b_out + l * C_, x, tmp, nullptr, MN_, C_, C_);
    layernorm_kernel<<<dim3(MN_ / 8), blk, 0, stream>>>(
        tmp, ln1_g + l * C_, ln1_b + l * C_, x, x16, MN_);

    // FFN1: h16 = relu(x16 @ W_ff1 + b_ff1)
    gemm_wmma_kernel<1, 0, 1><<<dim3(F_ / 64, MN_ / 128), blk, 0, stream>>>(
        x16, W1, b_ff1 + l * F_, nullptr, nullptr, h16, MN_, C_, F_);
    // FFN2 + residual: tmp = h16 @ W_ff2 + b_ff2 + x
    gemm_wmma_kernel<0, 1, 0><<<dim3(C_ / 64, MN_ / 128), blk, 0, stream>>>(
        h16, W2, b_ff2 + l * C_, x, tmp, nullptr, MN_, F_, C_);

    float* lnout = (l == L_ - 1) ? (float*)d_out : x;
    layernorm_kernel<<<dim3(MN_ / 8), blk, 0, stream>>>(
        tmp, ln2_g + l * C_, ln2_b + l * C_, lnout, (bf16*)nullptr, MN_);
  }
}
